// MyLSTM_1855425872429
// MI455X (gfx1250) — compile-verified, matrix operand order checked by reading, not run on hardware
//
#include <hip/hip_runtime.h>
#include <math.h>

// ---------------- types ----------------
typedef __bf16 v16bf __attribute__((ext_vector_type(16)));
typedef float  v8f   __attribute__((ext_vector_type(8)));
typedef unsigned int v4u __attribute__((ext_vector_type(4)));
typedef unsigned short bf16_t;   // storage-only bf16 (host-safe)

// ---------------- problem constants ----------------
// BATCH=250 (padded to 256), SEQ=200, EMB=100 (padded to 128), UNITS=512, 4U=2048
static constexpr int SEQ = 200;

// ---------------- workspace layout (bytes) ----------------
static constexpr size_t OFF_EMBBF = 0;                          // 100000 x 128 bf16 = 25,600,000
static constexpr size_t OFF_R0T   = 25600000;                   // 2048 x 512 bf16  =  2,097,152
static constexpr size_t OFF_K0T   = OFF_R0T  + 2097152;         // 2048 x 128 bf16  =    524,288
static constexpr size_t OFF_W1T   = OFF_K0T  + 524288;          // 2048 x 1024 bf16 =  4,194,304
static constexpr size_t OFF_HB0   = OFF_W1T  + 4194304;         // 256 x 1024 bf16  =    524,288
static constexpr size_t OFF_HB1   = OFF_HB0  + 524288;          // 256 x 1024 bf16
static constexpr size_t OFF_C0    = OFF_HB1  + 524288;          // 256 x 512 f32
static constexpr size_t OFF_C1    = OFF_C0   + 524288;          // 256 x 512 f32
static constexpr size_t OFF_H1F   = OFF_C1   + 524288;          // 256 x 512 f32
// total ~35.1 MB

// ---------------- device helpers ----------------
__device__ __forceinline__ bf16_t f2bf(float f) {
  unsigned u = __float_as_uint(f);
  u += 0x7FFFu + ((u >> 16) & 1u);   // round-to-nearest-even
  return (bf16_t)(u >> 16);
}

// Per-lane async copy of one 16-byte chunk: global -> LDS (ASYNCcnt-tracked).
__device__ __forceinline__ void async_copy16(unsigned ldsByteOff, const void* gsrc) {
  asm volatile("global_load_async_to_lds_b128 %0, %1, off"
               :: "v"(ldsByteOff), "v"(gsrc) : "memory");
}
__device__ __forceinline__ void async_wait0() {
  asm volatile("s_wait_asynccnt 0" ::: "memory");
}

// Load a 16-element bf16 WMMA fragment for this lane (global or LDS pointer).
// 16-bit A/B layout: lanes 0-15 hold K = [k..k+7, k+16..k+23],
// lanes 16-31 hold K = [k+8..k+15, k+24..k+31]  -> hiOff = (lane>>4)*8.
__device__ __forceinline__ v16bf load_frag(const bf16_t* rowBase, int kElem, int hiOff) {
  union { v16bf v; v4u q[2]; } u;
  u.q[0] = *reinterpret_cast<const v4u*>(rowBase + kElem + hiOff);
  u.q[1] = *reinterpret_cast<const v4u*>(rowBase + kElem + 16 + hiOff);
  return u.v;
}

__device__ __forceinline__ v8f wmma_bf16(v16bf a, v16bf b, v8f c) {
  // (neg_a, A, neg_b, B, c_mod, C, reuse_a, reuse_b)
  return __builtin_amdgcn_wmma_f32_16x16x32_bf16(false, a, false, b, (short)0, c, false, false);
}

__device__ __forceinline__ float sigm(float x) { return 1.0f / (1.0f + __expf(-x)); }

// ---------------- prep kernels ----------------
__global__ void zero_u32(unsigned int* __restrict__ p, int n) {
  int i = blockIdx.x * 256 + threadIdx.x;
  if (i < n) p[i] = 0u;
}

__global__ void cvt_emb(const float* __restrict__ emb, bf16_t* __restrict__ embBf) {
  int i = blockIdx.x * 256 + threadIdx.x;
  if (i >= 100000 * 128) return;
  int v = i >> 7, k = i & 127;
  embBf[i] = (k < 100) ? f2bf(emb[v * 100 + k]) : (bf16_t)0;
}

__global__ void cvt_r0(const float* __restrict__ r0, bf16_t* __restrict__ R0T) {
  int i = blockIdx.x * 256 + threadIdx.x;
  if (i >= 2048 * 512) return;
  int n = i >> 9, k = i & 511;
  R0T[i] = f2bf(r0[k * 2048 + n]);
}

__global__ void cvt_k0(const float* __restrict__ k0, bf16_t* __restrict__ K0T) {
  int i = blockIdx.x * 256 + threadIdx.x;
  if (i >= 2048 * 128) return;
  int n = i >> 7, k = i & 127;
  K0T[i] = (k < 100) ? f2bf(k0[k * 2048 + n]) : (bf16_t)0;
}

__global__ void cvt_w1(const float* __restrict__ k1, const float* __restrict__ r1,
                       bf16_t* __restrict__ W1T) {
  int i = blockIdx.x * 256 + threadIdx.x;
  if (i >= 2048 * 1024) return;
  int n = i >> 10, k = i & 1023;
  W1T[i] = (k < 512) ? f2bf(k1[k * 2048 + n]) : f2bf(r1[(k - 512) * 2048 + n]);
}

// ---------------- LSTM layer-0 step ----------------
// grid 64 x 256 -> 512 waves. Block: utile = blockIdx.x>>1 fixed; 8 waves cover
// mtiles (blockIdx.x&1)*8 .. +7. Weight slice (64 rows) async-staged to LDS once,
// shared by all 8 waves. K = 512 (h0@r0) + 128 (emb@k0, zero-padded).
__global__ __launch_bounds__(256) void lstm_step0(
    const bf16_t* __restrict__ hprev,   // [256][1024] bf16, h0 in cols 0..511 (prev step)
    bf16_t* __restrict__ hnext,         // [256][1024] bf16, write h0 cols 0..511
    const bf16_t* __restrict__ R0T,     // [2048][512]
    const bf16_t* __restrict__ K0T,     // [2048][128]
    const bf16_t* __restrict__ embBf,   // [100000][128]
    const int* __restrict__ tokens,     // [250][200]
    int t,
    const float* __restrict__ b0,       // [2048]
    float* __restrict__ c0)             // [256][512] f32
{
  extern __shared__ char smem[];                  // 81920 B: [0,64K) R slice, [64K,80K) K0 slice
  bf16_t* ldsR = (bf16_t*)smem;                   // 64 rows x 512 bf16
  bf16_t* ldsK = (bf16_t*)(smem + 65536);         // 64 rows x 128 bf16
  const unsigned ldsRBase = (unsigned)(uintptr_t)ldsR;
  const unsigned ldsKBase = (unsigned)(uintptr_t)ldsK;

  const int tid   = threadIdx.x;
  const int lane  = tid & 31;
  const int ll    = lane & 15;
  const int hiOff = (lane >> 4) << 3;
  const int utile = blockIdx.x >> 1;                       // 0..31
  const int mtile = ((blockIdx.x & 1) << 3) | (tid >> 5);  // 0..15
  const int n     = (utile << 4) | ll;                     // gate-local unit column 0..511
  const int mrow  = (mtile << 4) | ll;                     // A/token row

  // --- async stage: R0T slice rows {g*512 + utile*16 + c}, 64 rows x 1 KB = 4096 chunks
  #pragma unroll
  for (int i = 0; i < 16; ++i) {
    int q = tid + (i << 8);                  // 0..4095
    int row = q >> 6, inner = q & 63;        // 64 chunks per 512-elem row
    int g = row >> 4, c = row & 15;
    const bf16_t* src = R0T + (size_t)(((g << 9) + (utile << 4) + c) << 9) + (inner << 3);
    async_copy16(ldsRBase + (unsigned)(q << 4), src);
  }
  // --- async stage: K0T slice, 64 rows x 256 B = 1024 chunks
  #pragma unroll
  for (int i = 0; i < 4; ++i) {
    int q = tid + (i << 8);                  // 0..1023
    int row = q >> 4, inner = q & 15;        // 16 chunks per 128-elem row
    int g = row >> 4, c = row & 15;
    const bf16_t* src = K0T + (size_t)(((g << 9) + (utile << 4) + c) << 7) + (inner << 3);
    async_copy16(ldsKBase + (unsigned)(q << 4), src);
  }
  async_wait0();
  __syncthreads();

  v8f acc[4];
  #pragma unroll
  for (int g = 0; g < 4; ++g) {
    float bv = b0[(g << 9) + n];
    #pragma unroll
    for (int j = 0; j < 8; ++j) acc[g][j] = bv;
  }

  // recurrent part: h0_prev @ r0, K = 512 (B from LDS)
  const bf16_t* aRow = hprev + mrow * 1024;
  #pragma unroll 4
  for (int kc = 0; kc < 16; ++kc) {
    v16bf a = load_frag(aRow, kc << 5, hiOff);
    #pragma unroll
    for (int g = 0; g < 4; ++g) {
      const bf16_t* bRow = ldsR + (((g << 4) | ll) << 9);   // LDS row stride 512
      v16bf b = load_frag(bRow, kc << 5, hiOff);
      acc[g] = wmma_bf16(a, b, acc[g]);
    }
  }

  // input part: emb[token] @ k0, K padded 100 -> 128 (zeros in both operands)
  int tok = (mrow < 250) ? tokens[mrow * SEQ + t] : 0;
  const bf16_t* xRow = embBf + (size_t)tok * 128;
  #pragma unroll
  for (int kc = 0; kc < 4; ++kc) {
    v16bf a = load_frag(xRow, kc << 5, hiOff);
    #pragma unroll
    for (int g = 0; g < 4; ++g) {
      const bf16_t* bRow = ldsK + (((g << 4) | ll) << 7);   // LDS row stride 128
      v16bf b = load_frag(bRow, kc << 5, hiOff);
      acc[g] = wmma_bf16(a, b, acc[g]);
    }
  }

  // in-register LSTM cell update (C/D layout: M = j + 8*(lane>>4), N = lane&15)
  #pragma unroll
  for (int j = 0; j < 8; ++j) {
    int m = (mtile << 4) + hiOff + j;
    int idx = m * 512 + n;
    float cv = c0[idx];
    float iv = acc[0][j], fv = acc[1][j], cb = acc[2][j], ov = acc[3][j];
    float cn = sigm(fv) * cv + sigm(iv) * tanhf(cb);
    float hn = sigm(ov) * tanhf(cn);
    c0[idx] = cn;
    hnext[m * 1024 + n] = f2bf(hn);
  }
}

// ---------------- LSTM layer-1 step ----------------
// g1 = [h0_new | h1_prev] @ [k1; r1] + b1, K = 1024. W1T slice async-staged to LDS.
__global__ __launch_bounds__(256) void lstm_step1(
    const bf16_t* __restrict__ h0new,    // [256][1024] bf16, h0_t in cols 0..511
    const bf16_t* __restrict__ h1prev,   // [256][1024] bf16, h1_{t-1} in cols 512..1023
    bf16_t* __restrict__ h1next,         // [256][1024] bf16, write h1_t cols 512..1023
    const bf16_t* __restrict__ W1T,      // [2048][1024]
    const float* __restrict__ b1,        // [2048]
    float* __restrict__ c1,              // [256][512] f32
    float* __restrict__ h1f)             // [256][512] f32 (head input)
{
  extern __shared__ char smem[];                  // 131072 B: 64 rows x 1024 bf16
  bf16_t* ldsW = (bf16_t*)smem;
  const unsigned ldsWBase = (unsigned)(uintptr_t)ldsW;

  const int tid   = threadIdx.x;
  const int lane  = tid & 31;
  const int ll    = lane & 15;
  const int hiOff = (lane >> 4) << 3;
  const int utile = blockIdx.x >> 1;
  const int mtile = ((blockIdx.x & 1) << 3) | (tid >> 5);
  const int n     = (utile << 4) | ll;
  const int mrow  = (mtile << 4) | ll;

  // --- async stage: W1T slice rows {g*512 + utile*16 + c}, 64 rows x 2 KB = 8192 chunks
  #pragma unroll
  for (int i = 0; i < 32; ++i) {
    int q = tid + (i << 8);                  // 0..8191
    int row = q >> 7, inner = q & 127;       // 128 chunks per 1024-elem row
    int g = row >> 4, c = row & 15;
    const bf16_t* src = W1T + (size_t)(((g << 9) + (utile << 4) + c) << 10) + (inner << 3);
    async_copy16(ldsWBase + (unsigned)(q << 4), src);
  }
  async_wait0();
  __syncthreads();

  v8f acc[4];
  #pragma unroll
  for (int g = 0; g < 4; ++g) {
    float bv = b1[(g << 9) + n];
    #pragma unroll
    for (int j = 0; j < 8; ++j) acc[g][j] = bv;
  }

  const bf16_t* a0Row = h0new  + mrow * 1024;        // K 0..511
  const bf16_t* a1Row = h1prev + mrow * 1024 + 512;  // K 512..1023

  #pragma unroll 4
  for (int kc = 0; kc < 16; ++kc) {
    v16bf a = load_frag(a0Row, kc << 5, hiOff);
    #pragma unroll
    for (int g = 0; g < 4; ++g) {
      const bf16_t* bRow = ldsW + (((g << 4) | ll) << 10);  // LDS row stride 1024
      v16bf b = load_frag(bRow, kc << 5, hiOff);
      acc[g] = wmma_bf16(a, b, acc[g]);
    }
  }
  #pragma unroll 4
  for (int kc = 0; kc < 16; ++kc) {
    v16bf a = load_frag(a1Row, kc << 5, hiOff);
    #pragma unroll
    for (int g = 0; g < 4; ++g) {
      const bf16_t* bRow = ldsW + (((g << 4) | ll) << 10);
      v16bf b = load_frag(bRow, (16 + kc) << 5, hiOff);
      acc[g] = wmma_bf16(a, b, acc[g]);
    }
  }

  #pragma unroll
  for (int j = 0; j < 8; ++j) {
    int m = (mtile << 4) + hiOff + j;
    int idx = m * 512 + n;
    float cv = c1[idx];
    float iv = acc[0][j], fv = acc[1][j], cb = acc[2][j], ov = acc[3][j];
    float cn = sigm(fv) * cv + sigm(iv) * tanhf(cb);
    float hn = sigm(ov) * tanhf(cn);
    c1[idx] = cn;
    h1next[m * 1024 + 512 + n] = f2bf(hn);
    h1f[idx] = hn;
  }
}

// ---------------- dense head ----------------
// prob = sigmoid((h1 @ wd1 + bd1) @ wd2 + bd2)  -- tiny, one block.
__global__ void head_kernel(const float* __restrict__ h1f, const float* __restrict__ wd1,
                            const float* __restrict__ bd1, const float* __restrict__ wd2,
                            const float* __restrict__ bd2, float* __restrict__ out) {
  int b = threadIdx.x;
  if (b >= 250) return;
  float t0[32];
  #pragma unroll
  for (int j = 0; j < 32; ++j) t0[j] = bd1[j];
  for (int u = 0; u < 512; ++u) {
    float hv = h1f[b * 512 + u];
    #pragma unroll
    for (int j = 0; j < 32; ++j) t0[j] += hv * wd1[u * 32 + j];
  }
  float acc = bd2[0];
  #pragma unroll
  for (int j = 0; j < 32; ++j) acc += t0[j] * wd2[j];
  out[b] = 1.0f / (1.0f + __expf(-acc));
}

// ---------------- host launch ----------------
extern "C" void kernel_launch(void* const* d_in, const int* in_sizes, int n_in,
                              void* d_out, int out_size, void* d_ws, size_t ws_size,
                              hipStream_t stream) {
  (void)in_sizes; (void)n_in; (void)out_size; (void)ws_size;
  const int*   tokens = (const int*)  d_in[0];
  const float* emb    = (const float*)d_in[1];
  const float* k0     = (const float*)d_in[2];
  const float* r0     = (const float*)d_in[3];
  const float* b0     = (const float*)d_in[4];
  const float* k1     = (const float*)d_in[5];
  const float* r1     = (const float*)d_in[6];
  const float* b1     = (const float*)d_in[7];
  const float* wd1    = (const float*)d_in[8];
  const float* bd1    = (const float*)d_in[9];
  const float* wd2    = (const float*)d_in[10];
  const float* bd2    = (const float*)d_in[11];
  float* out = (float*)d_out;

  char* ws = (char*)d_ws;
  bf16_t* embBf = (bf16_t*)(ws + OFF_EMBBF);
  bf16_t* R0T   = (bf16_t*)(ws + OFF_R0T);
  bf16_t* K0T   = (bf16_t*)(ws + OFF_K0T);
  bf16_t* W1T   = (bf16_t*)(ws + OFF_W1T);
  bf16_t* hb[2] = { (bf16_t*)(ws + OFF_HB0), (bf16_t*)(ws + OFF_HB1) };
  float*  c0b   = (float*)(ws + OFF_C0);
  float*  c1b   = (float*)(ws + OFF_C1);
  float*  h1f   = (float*)(ws + OFF_H1F);

  // one-time prep (re-run every call for determinism)
  cvt_emb<<<(100000 * 128) / 256, 256, 0, stream>>>(emb, embBf);
  cvt_r0<<<(2048 * 512) / 256, 256, 0, stream>>>(r0, R0T);
  cvt_k0<<<(2048 * 128) / 256, 256, 0, stream>>>(k0, K0T);
  cvt_w1<<<(2048 * 1024) / 256, 256, 0, stream>>>(k1, r1, W1T);
  // zero hb0, hb1, c0, c1 (contiguous 2 MiB)
  zero_u32<<<(524288 + 255) / 256, 256, 0, stream>>>((unsigned int*)(ws + OFF_HB0), 524288);

  for (int t = 0; t < 200; ++t) {
    int p = t & 1;
    lstm_step0<<<64, 256, 81920, stream>>>(hb[p], hb[1 - p], R0T, K0T, embBf, tokens, t, b0, c0b);
    lstm_step1<<<64, 256, 131072, stream>>>(hb[1 - p], hb[p], hb[1 - p], W1T, b1, c1b, h1f);
  }

  head_kernel<<<1, 256, 0, stream>>>(h1f, wd1, bd1, wd2, bd2, out);
}
